// FingerprintPool_25752623907490
// MI455X (gfx1250) — matrix operand order repaired and checked by reading, not compiled, not run on hardware
//
#include <hip/hip_runtime.h>
#include <stdint.h>

typedef __attribute__((ext_vector_type(16))) _Float16 v16h;
typedef __attribute__((ext_vector_type(8)))  float    v8f;
typedef __attribute__((ext_vector_type(8)))  uint32_t v8u;

#define BB   64
#define AA   32
#define CC   64
#define LL   680          // 1 + 167 + 512
#define LLP  688          // padded to 43 full 16-row WMMA tiles
#define FPW  679          // fp columns (without the leading ones column)
#define NEGC (-1000000000000.0f)

// ---- dynamic LDS layout (bytes), all chunks 16B-aligned ----
#define OFF_XINNER 0
#define SZ_XINNER  (LLP * 65 * 4)       // 178880  x_inner[688][64] padded stride 65
#define OFF_XT     (OFF_XINNER + SZ_XINNER)
#define SZ_XT      (64 * 34 * 2)        // 4352    x^T f16 [c][a], padded stride 34
#define OFF_ATT    (OFF_XT + SZ_XT)
#define SZ_ATT     (8 * 16 * 34 * 2)    // 8704    per-wave attn tiles [16][32] pad 34
#define OFF_SSC    (OFF_ATT + SZ_ATT)
#define SZ_SSC     (96 * 4)             // 384     s[i][a] = dot(x[b,a,:], w_inner[i])
#define OFF_HITS   (OFF_SSC + SZ_SSC)
#define SZ_HITS    (LLP * 4)            // 2752
#define OFF_SC2    (OFF_HITS + SZ_HITS)
#define SZ_SC2     (LL * 4)             // 2720
#define OFF_WV     (OFF_SC2 + SZ_SC2)
#define SZ_WV      (LL * 4)             // 2720
#define OFF_RED    (OFF_WV + SZ_WV)
#define SZ_RED     (256 * 4)            // 1024
#define SMEM_BYTES (OFF_RED + SZ_RED)   // 201536 B  (< 320KB LDS per workgroup)

__global__ void __launch_bounds__(256)
fingerprint_pool_kernel(const float* __restrict__ x,
                        const int*   __restrict__ fp,
                        const float* __restrict__ w_inner,
                        const float* __restrict__ b_inner,
                        const float* __restrict__ w_inter,
                        const float* __restrict__ b_inter,
                        float*       __restrict__ out) {
  extern __shared__ __align__(16) unsigned char smem[];
  float*    xinner = (float*)   (smem + OFF_XINNER);
  _Float16* xT     = (_Float16*)(smem + OFF_XT);
  _Float16* attnT  = (_Float16*)(smem + OFF_ATT);
  float*    ssc    = (float*)   (smem + OFF_SSC);
  int*      hits   = (int*)     (smem + OFF_HITS);
  float*    sc2    = (float*)   (smem + OFF_SC2);
  float*    wv     = (float*)   (smem + OFF_WV);
  float*    red    = (float*)   (smem + OFF_RED);

  const int b    = blockIdx.x;
  const int tid  = threadIdx.x;
  const int wave = tid >> 5;
  const int lane = tid & 31;

  // ---------------- Phase 0: stage x^T (f16) + inner score dots (f32) ------
  for (int idx = tid; idx < AA * CC; idx += 256) {
    int a = idx >> 6, c = idx & 63;
    float v = x[(size_t)(b * AA + a) * CC + c];
    xT[c * 34 + a] = (_Float16)v;
  }
  if (tid < 96) {             // 3 segments x 32 rows
    int i = tid >> 5, a = tid & 31;
    const float* xr = x + (size_t)(b * AA + a) * CC;
    const float* wr = w_inner + i * CC;
    float acc = 0.f;
#pragma unroll 8
    for (int c = 0; c < CC; ++c) acc = fmaf(xr[c], wr[c], acc);
    ssc[i * AA + a] = acc;
  }
  __syncthreads();

  // ---------------- Phase 1: masked softmax over A + WMMA weighted sum -----
  _Float16* atile = attnT + wave * 16 * 34;
  const float bi0 = b_inner[0], bi1 = b_inner[1], bi2 = b_inner[2];
  const int*  fprow = fp + (size_t)(b * AA + lane) * FPW;

  for (int mt = wave; mt < 43; mt += 8) {          // 43 tiles of 16 l-rows
    // prefetch this lane's fp stretch for the *next* tile this wave will do
    {
      int nl = (mt + 8) * 16;
      if (nl < LL) __builtin_prefetch(fprow + (nl - 1), 0, 1);
    }
    for (int r = 0; r < 16; ++r) {
      int l   = mt * 16 + r;
      int seg = (l == 0) ? 0 : (l < 168 ? 1 : 2);
      float bi = (seg == 0) ? bi0 : (seg == 1 ? bi1 : bi2);
      int bit = 0;
      if (l == 0)        bit = 1;                           // ones column
      else if (l < LL)   bit = fprow[l - 1];
      float sc = bit ? (ssc[seg * AA + lane] + bi) : (bi + NEGC);
      // wave32 softmax over the A axis
      float mx = sc;
#pragma unroll
      for (int o = 16; o; o >>= 1) mx = fmaxf(mx, __shfl_xor(mx, o, 32));
      float e = expf(sc - mx);
      float sm = e;
#pragma unroll
      for (int o = 16; o; o >>= 1) sm += __shfl_xor(sm, o, 32);
      float w = bit ? (e / sm) : 0.f;   // exact zero for masked lanes (and all-zero rows)
      atile[r * 34 + lane] = (_Float16)w;
      if (lane == 0) {
        unsigned long long bal = __ballot(bit != 0);
        hits[l] = __popcll(bal);        // l < LLP always; rows >= LL never read
      }
    }
    __builtin_amdgcn_wave_barrier();
    asm volatile("s_wait_dscnt 0" ::: "memory");   // attn tile visible to all lanes

    // A fragment: 16-bit A 16x32 layout (ISA 7.12.2)
    int m  = lane & 15;
    int kb = (lane >> 4) << 3;                     // 0 or 8
    v8u au;
#pragma unroll
    for (int j = 0; j < 8; ++j) {
      int k = kb + 2 * (j & 3) + ((j >> 2) << 4);
      uint32_t t; __builtin_memcpy(&t, atile + m * 34 + k, 4);
      au[j] = t;
    }
    v16h afrag = __builtin_bit_cast(v16h, au);

    int n   = lane & 15;
    int kbB = (lane >> 4) << 3;                    // B: same K order as A (0 or 8)
    int mb  = (lane >> 4) << 3;
#pragma unroll
    for (int nt = 0; nt < 4; ++nt) {               // CC / 16 output column tiles
      v8u bu;
#pragma unroll
      for (int j = 0; j < 8; ++j) {
        int k = kbB + 2 * (j & 3) + ((j >> 2) << 4);
        uint32_t t; __builtin_memcpy(&t, xT + (nt * 16 + n) * 34 + k, 4);
        bu[j] = t;
      }
      v16h bfrag = __builtin_bit_cast(v16h, bu);
      v8f d = {};
      d = __builtin_amdgcn_wmma_f32_16x16x32_f16(
          /*neg_a=*/false, afrag, /*neg_b=*/false, bfrag,
          /*c_mod=*/(short)0, d, /*reuse_a=*/false, /*reuse_b=*/false);
      // D layout: VGPR v -> M = v + 8*(lane>=16), N = lane&15
      // xinner is padded to 688 rows: stores are always in-bounds (no exec masking)
      float* drow = xinner + (size_t)(mt * 16 + mb) * 65 + nt * 16 + n;
#pragma unroll
      for (int v = 0; v < 8; ++v) {
        drow[v * 65] = d[v];
      }
    }
  }
  __syncthreads();

  // ---------------- Phase 2: sc2[l] = dot(x_inner[l], w_inter) + mask ------
  for (int l = tid; l < LL; l += 256) {
    int seg = (l == 0) ? 0 : (l < 168 ? 1 : 2);
    const float* wr = w_inter + seg * CC;
    float acc = 0.f;
#pragma unroll 8
    for (int c = 0; c < CC; ++c) acc = fmaf(xinner[l * 65 + c], wr[c], acc);
    acc += b_inter[seg];
    if (hits[l] == 0) acc += NEGC;
    sc2[l] = acc;
  }
  __syncthreads();

  // ---------------- Phase 3: per-segment softmax over l ---------------------
  const int s_lo[3] = {0, 1, 168};
  const int s_hi[3] = {1, 168, 680};
  for (int sgi = 0; sgi < 3; ++sgi) {
    int s0 = s_lo[sgi], s1 = s_hi[sgi];
    float mx = -3.4e38f;
    for (int l = s0 + tid; l < s1; l += 256) mx = fmaxf(mx, sc2[l]);
    red[tid] = mx; __syncthreads();
    for (int st = 128; st; st >>= 1) {
      if (tid < st) red[tid] = fmaxf(red[tid], red[tid + st]);
      __syncthreads();
    }
    float segmax = red[0]; __syncthreads();
    float sm = 0.f;
    for (int l = s0 + tid; l < s1; l += 256) {
      float e = expf(sc2[l] - segmax);
      wv[l] = e; sm += e;
    }
    red[tid] = sm; __syncthreads();
    for (int st = 128; st; st >>= 1) {
      if (tid < st) red[tid] += red[tid + st];
      __syncthreads();
    }
    float inv = 1.f / red[0]; __syncthreads();
    for (int l = s0 + tid; l < s1; l += 256) wv[l] *= inv;
    __syncthreads();
  }

  // ---------------- Phase 4: out[b,c] = sum_l wv[l] * x_inner[l,c] ----------
  int c = tid & 63, part = tid >> 6;   // 4 partial accumulators per column
  float acc = 0.f;
  for (int l = part; l < LL; l += 4) acc = fmaf(wv[l], xinner[l * 65 + c], acc);
  red[part * 64 + c] = acc;
  __syncthreads();
  if (tid < CC)
    out[(size_t)b * CC + tid] =
        red[tid] + red[64 + tid] + red[128 + tid] + red[192 + tid];
}

extern "C" void kernel_launch(void* const* d_in, const int* in_sizes, int n_in,
                              void* d_out, int out_size, void* d_ws, size_t ws_size,
                              hipStream_t stream) {
  const float* x       = (const float*)d_in[0];
  // d_in[1] = batch (unused: rows are already grouped B x A)
  const int*   fp      = (const int*)  d_in[2];
  // d_in[3] = fp_length (compile-time constants here)
  const float* w_inner = (const float*)d_in[4];
  const float* b_inner = (const float*)d_in[5];
  const float* w_inter = (const float*)d_in[6];
  const float* b_inter = (const float*)d_in[7];
  float* out = (float*)d_out;

  fingerprint_pool_kernel<<<BB, 256, SMEM_BYTES, stream>>>(
      x, fp, w_inner, b_inner, w_inter, b_inter, out);
}
